// EIRNN_91096256348416
// MI455X (gfx1250) — matrix operand order repaired and checked by reading, not compiled, not run on hardware
//
#include <hip/hip_runtime.h>

// ---------------- problem constants ----------------
#define BB    256
#define TT    512
#define II    128
#define HH    512
#define OO    64
#define ALPHA 0.2f
#define BETA  8.0f
#define THR   20.0f

#define KRES  256               // K range [0,KRES) of Whh resident in LDS
#define NTHR  512               // 16 waves per workgroup

// ---------------- WMMA types ----------------
typedef __attribute__((ext_vector_type(16))) __bf16         v16bf;
typedef __attribute__((ext_vector_type(8)))  float          v8f;
typedef __attribute__((ext_vector_type(8)))  unsigned short v8u;

union FragCvt { v8u half2[2]; v16bf bf; };

__device__ __forceinline__ v16bf make_frag(v8u lo, v8u hi) {
    FragCvt f;
    f.half2[0] = lo;
    f.half2[1] = hi;
    return f.bf;
}

__device__ __forceinline__ v8f wmma_bf16(v16bf a, v16bf b, v8f c) {
    // (neg_a, A, neg_b, B, c_mod, C, reuse_a, reuse_b)
    return __builtin_amdgcn_wmma_f32_16x16x32_bf16(false, a, false, b,
                                                   (short)0, c, false, false);
}

// round-to-nearest-even f32 -> bf16 bits
__device__ __forceinline__ unsigned short f2bf(float f) {
    unsigned int u = __builtin_bit_cast(unsigned int, f);
    return (unsigned short)((u + 0x7FFFu + ((u >> 16) & 1u)) >> 16);
}

// A-fragment (16x32 bf16, M x K) from a row-major row pointer (row m, +kt*32).
// K(e) = e + 16*(lane>=16)  ->  two contiguous 16B LDS loads.
__device__ __forceinline__ v16bf a_frag_row(const unsigned short* row, int hi) {
    v8u lo = *(const v8u*)(row + hi * 16);
    v8u hv = *(const v8u*)(row + hi * 16 + 8);
    return make_frag(lo, hv);
}

// B-fragment (32x16 bf16, K x N) where B[k,n] = W[n][k] (row-major W, K contig).
// p = &W[n*rowlen + kt*32 + 16*(lane>=16)];  K(e) = e (contiguous).
__device__ __forceinline__ v16bf b_frag_row(const unsigned short* p) {
    v8u lo = *(const v8u*)(p);
    v8u hv = *(const v8u*)(p + 8);
    return make_frag(lo, hv);
}

// ---------------- prep: cast weights + x to bf16 in workspace ----------------
#define N_WXH  (HH * II)               // 65536
#define N_WHH  (HH * HH)               // 262144
#define N_WOUT (OO * HH)               // 32768
#define N_X    ((size_t)BB * TT * II)  // 16777216
#define N_TOT  (N_WXH + N_WHH + N_WOUT + N_X)

__global__ void cast_bf16_kernel(const float* __restrict__ Wxh,
                                 const float* __restrict__ Whh,
                                 const float* __restrict__ Wout,
                                 const float* __restrict__ x,
                                 unsigned short* __restrict__ ws) {
    size_t i = (size_t)blockIdx.x * blockDim.x + threadIdx.x;
    if (i >= N_TOT) return;
    float v;
    if (i < N_WXH)                       v = Wxh[i];
    else if (i < N_WXH + N_WHH)          v = Whh[i - N_WXH];
    else if (i < N_WXH + N_WHH + N_WOUT) v = Wout[i - N_WXH - N_WHH];
    else                                 v = x[i - N_WXH - N_WHH - N_WOUT];
    ws[i] = f2bf(v);
}

// ---------------- persistent scan kernel ----------------
// grid = 16 workgroups (one 16-row batch tile each), 512 threads = 16 waves
// (4 waves/SIMD). Wave w owns H columns [32w, 32w+32) (2 N-tiles). The K<256
// half of Whh is LDS-resident (256 KB); K>=256 streams from L2 (prefetched at
// step start). h state: f32 in registers, bf16 mirror in LDS.
__launch_bounds__(NTHR)
__global__ void eirnn_scan_kernel(const float* __restrict__ b_h,
                                  const float* __restrict__ b_out,
                                  const unsigned short* __restrict__ Wxh,   // [H][I] bf16
                                  const unsigned short* __restrict__ Whh,   // [H][H] bf16
                                  const unsigned short* __restrict__ Wout,  // [O][H] bf16
                                  const unsigned short* __restrict__ xbf,   // [B][T][I] bf16
                                  float* __restrict__ out) {                // [B][T][O] f32
    __shared__ unsigned short whh_l[HH * KRES];  // 256 KB: Whh[:, 0:256] bf16
    __shared__ unsigned short hs[16 * HH];       //  16 KB: bf16 h (A-matrix source)
    __shared__ unsigned short xs[16 * II];       //   4 KB: x_t tile as bf16

    const int tid  = threadIdx.x;
    const int lane = tid & 31;
    const int wave = tid >> 5;      // 0..15
    const int m    = lane & 15;     // row / N index within 16x16 tile
    const int hi   = lane >> 4;     // lane half
    const int bb   = blockIdx.x;    // batch tile (16 rows)
    const int colbase = wave * 32;  // this wave's H columns
    const int n0 = colbase + m;
    const int n1 = colbase + 16 + m;

    // ---- one-time: stage Whh[:, 0:KRES) into LDS (row-major, K contiguous) ----
    for (int j = tid; j < HH * KRES / 8; j += NTHR) {       // v8u chunks
        const int n    = j >> 5;          // 32 chunks per row
        const int koff = (j & 31) * 8;
        *(v8u*)(whh_l + n * KRES + koff) =
            *(const v8u*)(Whh + (size_t)n * HH + koff);
    }

    // biases (constant over t)
    const float bh0 = b_h[n0];
    const float bh1 = b_h[n1];
    const float bo  = (wave < 4) ? b_out[wave * 16 + m] : 0.0f;

    // streamed-half row pointers (t-invariant)
    const unsigned short* ws0 = Whh + (size_t)n0 * HH + KRES + hi * 16;
    const unsigned short* ws1 = Whh + (size_t)n1 * HH + KRES + hi * 16;

    // f32 recurrent state: element (nt,r) = h[ M=r+8*hi ][ N=colbase+nt*16+m ]
    float hreg[2][8];
#pragma unroll
    for (int nt = 0; nt < 2; ++nt)
#pragma unroll
        for (int r = 0; r < 8; ++r) hreg[nt][r] = 0.0f;

    for (int i = tid; i < 16 * HH; i += NTHR) hs[i] = 0;  // bf16 zero
    __syncthreads();

    for (int t = 0; t < TT; ++t) {
        // ---- stage x_t tile (16 rows x 128) into LDS ----
        if (tid < 256) {
            const int mr = tid >> 4;
            const int i  = (tid & 15) * 8;
            const v8u xv = *(const v8u*)(xbf + (((size_t)(bb * 16 + mr) * TT + t) * II + i));
            *(v8u*)(xs + mr * II + i) = xv;
        }
        // warm the streamed Whh rows while we do x-proj + LDS-resident half
        __builtin_prefetch(ws0, 0, 0);
        __builtin_prefetch(ws0 + 128, 0, 0);
        __builtin_prefetch(ws1, 0, 0);
        __builtin_prefetch(ws1 + 128, 0, 0);
        __syncthreads();  // bar A: xs ready; prior-step hs readers done

        v8f acc0, acc1;
#pragma unroll
        for (int r = 0; r < 8; ++r) { acc0[r] = bh0; acc1[r] = bh1; }

        // ---- input projection: K = 128 (4 k-tiles), Wxh fragments t-invariant ----
        {
            const unsigned short* w0 = Wxh + (size_t)n0 * II + hi * 16;
            const unsigned short* w1 = Wxh + (size_t)n1 * II + hi * 16;
#pragma unroll
            for (int kt = 0; kt < 4; ++kt) {
                v16bf a  = a_frag_row(xs + m * II + kt * 32, hi);
                acc0 = wmma_bf16(a, b_frag_row(w0 + kt * 32), acc0);
                acc1 = wmma_bf16(a, b_frag_row(w1 + kt * 32), acc1);
            }
        }
        __builtin_amdgcn_sched_barrier(0);

        // ---- recurrence, LDS-resident half: K in [0,256) ----
        {
            const unsigned short* w0 = whh_l + n0 * KRES + hi * 16;
            const unsigned short* w1 = whh_l + n1 * KRES + hi * 16;
#pragma unroll 2
            for (int kt = 0; kt < 8; ++kt) {
                v16bf a  = a_frag_row(hs + m * HH + kt * 32, hi);
                acc0 = wmma_bf16(a, b_frag_row(w0 + kt * 32), acc0);
                acc1 = wmma_bf16(a, b_frag_row(w1 + kt * 32), acc1);
            }
        }
        __builtin_amdgcn_sched_barrier(0);

        // ---- recurrence, streamed half: K in [256,512) ----
        {
#pragma unroll 2
            for (int kt = 0; kt < 8; ++kt) {
                v16bf a = a_frag_row(hs + m * HH + KRES + kt * 32, hi);
                acc0 = wmma_bf16(a, b_frag_row(ws0 + kt * 32), acc0);
                acc1 = wmma_bf16(a, b_frag_row(ws1 + kt * 32), acc1);
            }
        }
        __builtin_amdgcn_sched_barrier(0);

        __syncthreads();  // bar B: all hs reads finished before overwrite

        // ---- h = (1-ALPHA)*h + ALPHA*phi(pre); publish bf16 h ----
#pragma unroll
        for (int nt = 0; nt < 2; ++nt)
#pragma unroll
            for (int r = 0; r < 8; ++r) {
                const float pre = (nt == 0) ? acc0[r] : acc1[r];
                const float z   = BETA * pre;
                const float zc  = fminf(z, THR);
                const float sp  = __logf(1.0f + __expf(zc)) * (1.0f / BETA);
                const float ph  = (z > THR) ? pre : sp;
                const float hn  = (1.0f - ALPHA) * hreg[nt][r] + ALPHA * ph;
                hreg[nt][r] = hn;
                const int mrow = r + hi * 8;
                const int ncol = colbase + nt * 16 + m;
                hs[mrow * HH + ncol] = f2bf(hn);
            }
        __syncthreads();  // bar C: new hs visible

        // ---- readout y = h @ Wout^T + b_out (waves 0..3, one O n-tile each) ----
        if (wave < 4) {
            v8f yacc;
#pragma unroll
            for (int r = 0; r < 8; ++r) yacc[r] = bo;
            const int n = wave * 16 + m;  // O column
            const unsigned short* wop = Wout + (size_t)n * HH + hi * 16;
#pragma unroll 2
            for (int kt = 0; kt < 16; ++kt) {
                v16bf a = a_frag_row(hs + m * HH + kt * 32, hi);
                yacc = wmma_bf16(a, b_frag_row(wop + kt * 32), yacc);
            }
#pragma unroll
            for (int r = 0; r < 8; ++r) {
                const int mrow = r + hi * 8;
                out[((size_t)(bb * 16 + mrow) * TT + t) * OO + n] = yacc[r];
            }
        }
    }
}

// ---------------- host entry ----------------
extern "C" void kernel_launch(void* const* d_in, const int* in_sizes, int n_in,
                              void* d_out, int out_size, void* d_ws, size_t ws_size,
                              hipStream_t stream) {
    const float* x    = (const float*)d_in[0];  // [B,T,I]
    const float* Wxh  = (const float*)d_in[1];  // [H,I]
    const float* Whh  = (const float*)d_in[2];  // [H,H]
    const float* bh   = (const float*)d_in[3];  // [H]
    const float* Wout = (const float*)d_in[4];  // [O,H]
    const float* bout = (const float*)d_in[5];  // [O]
    float* out = (float*)d_out;                 // [B,T,O]

    unsigned short* ws      = (unsigned short*)d_ws;
    unsigned short* wxh_bf  = ws;
    unsigned short* whh_bf  = wxh_bf + N_WXH;
    unsigned short* wout_bf = whh_bf + N_WHH;
    unsigned short* x_bf    = wout_bf + N_WOUT;

    const int cast_blocks = (int)((N_TOT + 255) / 256);
    cast_bf16_kernel<<<cast_blocks, 256, 0, stream>>>(Wxh, Whh, Wout, x, ws);

    eirnn_scan_kernel<<<BB / 16, NTHR, 0, stream>>>(bh, bout, wxh_bf, whh_bf,
                                                    wout_bf, x_bf, out);
}
